// MambaOperator_31568009626255
// MI455X (gfx1250) — matrix operand order repaired and checked by reading, not compiled, not run on hardware
//
#include <hip/hip_runtime.h>
#include <hip/hip_bf16.h>

// ---------------------------------------------------------------------------
// Mamba block for MI455X (gfx1250, wave32, WMMA + Tensor Data Mover).
// Sizes: B=4, L=2048, D_MODEL=1024, D_STATE=16, D_INNER=2048, DT_RANK=64.
// ---------------------------------------------------------------------------

typedef __bf16 bf16_t;
typedef __attribute__((ext_vector_type(16))) __bf16 v16bf;
typedef __attribute__((ext_vector_type(8)))  __bf16 v8bf;
typedef __attribute__((ext_vector_type(4)))  __bf16 v4bf;
typedef __attribute__((ext_vector_type(8)))  float  v8f;
typedef unsigned int u32x4 __attribute__((ext_vector_type(4)));
typedef int          i32x4 __attribute__((ext_vector_type(4)));
typedef int          i32x8 __attribute__((ext_vector_type(8)));

#define LDS_STRIDE 48   // bf16 elems per LDS row; 96B rows keep b128 LDS reads 16B-aligned

#if defined(__gfx1250__) && __has_builtin(__builtin_amdgcn_tensor_load_to_lds) && \
    __has_builtin(__builtin_amdgcn_s_wait_tensorcnt)
#define USE_TDM 1
#else
#define USE_TDM 0
#endif

__device__ __forceinline__ float silu_f(float x) {
    return x / (1.f + __expf(-x));
}
__device__ __forceinline__ float softplus_f(float x) {
    return (x > 20.f) ? x : __logf(1.f + __expf(x));
}

#if USE_TDM
// ---------------------------------------------------------------------------
// TDM: async-copy a tile_h x tile_w rectangle of bf16 (row stride tensor_w
// elems) from global into LDS, padding each 64B LDS row by 32B so the LDS
// row stride is 96B (= LDS_STRIDE bf16). D# layout per CDNA5 ISA ch.8.
// This toolchain exposes the 6-arg builtin: (g0, g1, g2, g3, g4, cpol).
// ---------------------------------------------------------------------------
__device__ __forceinline__ void tdm_load_tile_bf16(
    const bf16_t* gtile, unsigned lds_byte_off,
    unsigned tensor_w, unsigned tensor_h,
    unsigned tile_w,   unsigned tile_h)
{
    unsigned long long ga = (unsigned long long)(size_t)gtile;
    u32x4 g0 = {};
    g0[0] = 1u;                                               // count=1 (valid D#)
    g0[1] = lds_byte_off;                                     // lds_addr
    g0[2] = (unsigned)ga;                                     // global_addr[31:0]
    g0[3] = (unsigned)((ga >> 32) & 0x1FFFFFFull) | (2u << 30); // addr[56:32] | type=2

    unsigned long long q0 =
          ((unsigned long long)1u << 16)                      // data_size = 1 (2 bytes)
        | ((unsigned long long)1u << 20)                      // pad_enable
        | ((unsigned long long)3u << 22)                      // pad_interval: 16 DW = 64B
        | ((unsigned long long)7u << 25)                      // pad_amount:    8 DW = 32B
        | ((unsigned long long)(tensor_w & 0xFFFFu) << 48);   // tensor_dim0[15:0]
    unsigned long long q1 =
          ((unsigned long long)(tensor_w >> 16))              // tensor_dim0[31:16]
        | ((unsigned long long)tensor_h << 16)                // tensor_dim1
        | ((unsigned long long)(tile_w & 0xFFFFu) << 48);     // tile_dim0
    unsigned long long q2 =
          ((unsigned long long)(tile_h & 0xFFFFu))            // tile_dim1 (tile_dim2=0)
        | ((unsigned long long)tensor_w << 32);               // tensor_dim0_stride[31:0]
    unsigned long long q3 = 0;                                // stride hi / dim1_stride = 0

    i32x8 g1 = {};
    g1[0] = (int)q0; g1[1] = (int)(q0 >> 32);
    g1[2] = (int)q1; g1[3] = (int)(q1 >> 32);
    g1[4] = (int)q2; g1[5] = (int)(q2 >> 32);
    g1[6] = (int)q3; g1[7] = (int)(q3 >> 32);
    i32x4 gz4 = {0, 0, 0, 0};
    i32x8 gz8 = {0, 0, 0, 0, 0, 0, 0, 0};
    __builtin_amdgcn_tensor_load_to_lds(g0, g1, gz4, gz4, gz8, 0);
}
#endif

// ---------------------------------------------------------------------------
// One-time prepass: W[K,N] fp32 row-major -> Wt[N,K] bf16 row-major.
// Makes GEMM B-tiles contiguous-in-K so the TDM can stage them directly.
// ---------------------------------------------------------------------------
__global__ __launch_bounds__(256) void cvt_transpose_kernel(
    const float* __restrict__ W, bf16_t* __restrict__ Wt, int K, int N)
{
    size_t idx = (size_t)blockIdx.x * 256 + threadIdx.x;    // over K*N
    int n = (int)(idx % N);
    int k = (int)(idx / N);
    Wt[(size_t)n * K + k] = (bf16_t)W[idx];
}

// ---------------------------------------------------------------------------
// WMMA GEMM:  C[M,N] = A[M,K] (fp32 row-major) * Wt[N,K]^T (bf16, pre-transposed)
// Block: 256 threads (8 wave32), tile 128 x (BNT*16); each wave owns a
// 16 x (BNT*16) strip -> BNT v_wmma_f32_16x16x32_bf16 per K-step of 32.
// B tile staged Global->LDS by the Tensor Data Mover (padded to 96B rows).
// AMODE: 0 = plain A; 1 = A := A * silu(A2)
// EMODE: 0 = plain store; 1 = softplus(v + bias[col]); 2 = split cols at Nhalf
// ---------------------------------------------------------------------------
template <int BNT, int AMODE, int EMODE>
__global__ __launch_bounds__(256) void gemm_wmma(
    const float* __restrict__ A, const float* __restrict__ A2,
    const bf16_t* __restrict__ Bw, const float* __restrict__ bias,
    float* __restrict__ C, float* __restrict__ C2,
    int M, int N, int K, int lda, int ldb, int ldc, int Nhalf)
{
    constexpr int BM = 128;
    constexpr int BK = 32;
    constexpr int BN = BNT * 16;

    __shared__ alignas(16) bf16_t As[BM * LDS_STRIDE];   // 128x32 A tile (bf16)
    __shared__ alignas(16) bf16_t Bst[BN * LDS_STRIDE];  // BN x 32 B tile (K-contig rows)

    const int tid  = threadIdx.x;
    const int wave = tid >> 5;
    const int lane = tid & 31;
    const int m0   = blockIdx.y * BM;
    const int n0   = blockIdx.x * BN;

    v8f acc[BNT] = {};

    const int rowA  = wave * 16 + (lane & 15);
    const int kA    = (lane < 16) ? 0 : 8;    // ISA 16-bit A 16x32 lane layout
    const int kB    = (lane < 16) ? 0 : 8;    // ISA 16-bit B 32x16 lane layout
    const int laneN = lane & 15;

    for (int k0 = 0; k0 < K; k0 += BK) {
        // ---- stage B tile via Tensor Data Mover (wave 0 issues + waits) ----
#if USE_TDM
        if (wave == 0) {
            tdm_load_tile_bf16(Bw + (size_t)n0 * ldb + k0,
                               (unsigned)(size_t)&Bst[0],
                               (unsigned)ldb, (unsigned)(N - n0),
                               BK, BN);
            __builtin_amdgcn_s_wait_tensorcnt(0);
        }
#else
        for (int idx = tid; idx < BN * 4; idx += 256) {
            int n  = idx >> 2;
            int kc = (idx & 3) * 8;
            v8bf v = *(const v8bf*)(Bw + (size_t)(n0 + n) * ldb + k0 + kc);
            *(v8bf*)&Bst[n * LDS_STRIDE + kc] = v;
        }
#endif
        // ---- stage A tile (128 x 32 fp32 -> bf16), 4 float4 per thread ----
        #pragma unroll
        for (int i = 0; i < 4; ++i) {
            int idx = tid + i * 256;                 // 0..1023
            int r   = idx >> 3;                      // 0..127
            int c4  = (idx & 7) << 2;                // 0..28
            const float* src = A + (size_t)(m0 + r) * lda + k0 + c4;
            float4 v = *(const float4*)src;
            if (AMODE == 1) {
                float4 g = *(const float4*)(A2 + (size_t)(m0 + r) * lda + k0 + c4);
                v.x *= silu_f(g.x); v.y *= silu_f(g.y);
                v.z *= silu_f(g.z); v.w *= silu_f(g.w);
            }
            __builtin_prefetch(src + BK, 0, 1);      // speculative global_prefetch_b8
            v4bf pv = { (bf16_t)v.x, (bf16_t)v.y, (bf16_t)v.z, (bf16_t)v.w };
            *(v4bf*)&As[r * LDS_STRIDE + c4] = pv;   // one ds_store_b64
        }
        __syncthreads();

        // ---- A fragment: two aligned 16B LDS reads per lane ----
        v8bf a0 = *(const v8bf*)&As[rowA * LDS_STRIDE + kA];
        v8bf a1 = *(const v8bf*)&As[rowA * LDS_STRIDE + kA + 16];
        v16bf afrag = __builtin_shufflevector(a0, a1,
            0,1,2,3,4,5,6,7,8,9,10,11,12,13,14,15);

        // ---- B fragments + WMMA ----
        #pragma unroll
        for (int t = 0; t < BNT; ++t) {
            int col = t * 16 + laneN;
            v8bf b0 = *(const v8bf*)&Bst[col * LDS_STRIDE + kB];
            v8bf b1 = *(const v8bf*)&Bst[col * LDS_STRIDE + kB + 16];
            v16bf bfrag = __builtin_shufflevector(b0, b1,
                0,1,2,3,4,5,6,7,8,9,10,11,12,13,14,15);
            acc[t] = __builtin_amdgcn_wmma_f32_16x16x32_bf16(
                false, afrag, false, bfrag, (short)0, acc[t], false, false);
        }
        __syncthreads();
    }

    // ---- epilogue: C/D layout: VGPR i -> M = i + (lane<16 ? 0 : 8), N = lane&15 ----
    const int rbase = m0 + wave * 16 + ((lane < 16) ? 0 : 8);
    #pragma unroll
    for (int t = 0; t < BNT; ++t) {
        int col = n0 + t * 16 + laneN;
        #pragma unroll
        for (int i = 0; i < 8; ++i) {
            int   row = rbase + i;
            float v   = acc[t][i];
            if (EMODE == 1) {
                C[(size_t)row * ldc + col] = softplus_f(v + bias[col]);
            } else if (EMODE == 2) {
                if (col < Nhalf) C [(size_t)row * ldc + col]           = v;
                else             C2[(size_t)row * ldc + (col - Nhalf)] = v;
            } else {
                C[(size_t)row * ldc + col] = v;
            }
        }
    }
}

// ---------------------------------------------------------------------------
// Depthwise causal conv (k=4) along L + bias + SiLU.  x layout [B, L, D_INNER].
// ---------------------------------------------------------------------------
__global__ __launch_bounds__(256) void conv_silu_kernel(
    const float* __restrict__ x, const float* __restrict__ w,
    const float* __restrict__ b, float* __restrict__ out)
{
    const int L = 2048, D = 2048;
    size_t idx = (size_t)blockIdx.x * 256 + threadIdx.x;   // over B*L*D = 2^24
    int d  = (int)(idx & (D - 1));
    size_t bl = idx >> 11;
    int l  = (int)(bl & (L - 1));
    int bb = (int)(bl >> 11);

    const float* xb = x + ((size_t)bb * L) * D + d;
    float acc = b[d];
    #pragma unroll
    for (int j = 0; j < 4; ++j) {
        int t = l - 3 + j;
        if (t >= 0) acc += w[d * 4 + j] * xb[(size_t)t * D];
    }
    out[idx] = silu_f(acc);
}

// ---------------------------------------------------------------------------
// Selective scan: one lane per (b,d) channel, h[16] resident in VGPRs.
// B/C rows broadcast through LDS; y = sum_n h[n]*C[n] + D[d]*u.
// ---------------------------------------------------------------------------
__global__ __launch_bounds__(256) void scan_kernel(
    const float* __restrict__ delta, const float* __restrict__ dbc,
    const float* __restrict__ xs,    const float* __restrict__ hiddens,
    const float* __restrict__ A_log, const float* __restrict__ Dp,
    float* __restrict__ y)
{
    const int L = 2048, D = 2048, NS = 16, LDB = 96;
    int bb = blockIdx.x >> 3;
    int d  = (blockIdx.x & 7) * 256 + threadIdx.x;

    float Ar[NS], h[NS];
    #pragma unroll
    for (int n = 0; n < NS; ++n) {
        Ar[n] = -__expf(A_log[(size_t)d * NS + n]);
        h[n]  = hiddens[((size_t)bb * D + d) * NS + n];
    }
    const float Dd = Dp[d];

    __shared__ float sB[NS], sC[NS];

    for (int t = 0; t < L; ++t) {
        size_t row = (size_t)bb * L + t;
        if (threadIdx.x < NS)          sB[threadIdx.x]      = dbc[row * LDB + 64 + threadIdx.x];
        else if (threadIdx.x < 2 * NS) sC[threadIdx.x - NS] = dbc[row * LDB + 80 + (threadIdx.x - NS)];
        __syncthreads();

        float dt  = delta[row * D + d];
        float u   = xs[row * D + d];
        float dtu = dt * u;
        float acc = Dd * u;
        #pragma unroll
        for (int n = 0; n < NS; ++n) {
            h[n] = __expf(dt * Ar[n]) * h[n] + dtu * sB[n];
            acc += h[n] * sC[n];
        }
        y[row * D + d] = acc;
        __syncthreads();
    }
}

// ---------------------------------------------------------------------------
extern "C" void kernel_launch(void* const* d_in, const int* in_sizes, int n_in,
                              void* d_out, int out_size, void* d_ws, size_t ws_size,
                              hipStream_t stream)
{
    const float* u         = (const float*)d_in[0];
    const float* hiddens   = (const float*)d_in[1];
    const float* in_proj_w = (const float*)d_in[2];
    const float* conv_w    = (const float*)d_in[3];
    const float* conv_b    = (const float*)d_in[4];
    const float* x_proj_w  = (const float*)d_in[5];
    const float* dt_proj_w = (const float*)d_in[6];
    const float* dt_proj_b = (const float*)d_in[7];
    const float* A_log     = (const float*)d_in[8];
    const float* Dp        = (const float*)d_in[9];
    const float* out_proj_w= (const float*)d_in[10];

    const int M = 8192;                 // B*L
    const size_t MLD = (size_t)M * 2048;

    float* ws    = (float*)d_ws;
    float* x_pre = ws;                          // [M, 2048]
    float* res   = x_pre + MLD;                 // [M, 2048]
    float* xs    = res   + MLD;                 // [M, 2048]
    float* dbc   = xs    + MLD;                 // [M, 96]
    float* delta = dbc   + (size_t)M * 96;      // [M, 2048]
    float* yb    = delta + MLD;                 // [M, 2048]

    // bf16 transposed weights (prepass output)
    bf16_t* wt_in  = (bf16_t*)(yb + MLD);            // [4096, 1024]
    bf16_t* wt_x   = wt_in  + (size_t)4096 * 1024;   // [96, 2048]
    bf16_t* wt_dt  = wt_x   + (size_t)96 * 2048;     // [2048, 64]
    bf16_t* wt_out = wt_dt  + (size_t)2048 * 64;     // [1024, 2048]

    dim3 blk(256);

    // 0) weight prepass: fp32 [K,N] -> bf16 [N,K]
    cvt_transpose_kernel<<<dim3(1024u * 4096u / 256u), blk, 0, stream>>>(in_proj_w, wt_in, 1024, 4096);
    cvt_transpose_kernel<<<dim3(2048u * 96u   / 256u), blk, 0, stream>>>(x_proj_w,  wt_x,  2048, 96);
    cvt_transpose_kernel<<<dim3(64u   * 2048u / 256u), blk, 0, stream>>>(dt_proj_w, wt_dt, 64,   2048);
    cvt_transpose_kernel<<<dim3(2048u * 1024u / 256u), blk, 0, stream>>>(out_proj_w,wt_out,2048, 1024);

    // 1) in_proj: [8192,1024] x [1024,4096] -> split x / res
    gemm_wmma<4, 0, 2><<<dim3(4096 / 64, M / 128), blk, 0, stream>>>(
        u, nullptr, wt_in, nullptr, x_pre, res,
        M, 4096, 1024, 1024, 1024, 2048, 2048);

    // 2) depthwise causal conv + SiLU
    conv_silu_kernel<<<dim3((unsigned)(((size_t)M * 2048) / 256)), blk, 0, stream>>>(
        x_pre, conv_w, conv_b, xs);

    // 3) x_proj: [8192,2048] x [2048,96] -> dbc
    gemm_wmma<6, 0, 0><<<dim3(1, M / 128), blk, 0, stream>>>(
        xs, nullptr, wt_x, nullptr, dbc, nullptr,
        M, 96, 2048, 2048, 2048, 96, 0);

    // 4) dt_proj + bias + softplus: [8192,64] x [64,2048] -> delta
    gemm_wmma<4, 0, 1><<<dim3(2048 / 64, M / 128), blk, 0, stream>>>(
        dbc, nullptr, wt_dt, dt_proj_b, delta, nullptr,
        M, 2048, 64, 96, 64, 2048, 0);

    // 5) selective scan (+ x*D folded in)
    scan_kernel<<<dim3(32), blk, 0, stream>>>(
        delta, dbc, xs, hiddens, A_log, Dp, yb);

    // 6) out_proj with fused gate: (y * silu(res)) x [2048,1024] -> d_out
    gemm_wmma<4, 1, 0><<<dim3(1024 / 64, M / 128), blk, 0, stream>>>(
        yb, res, wt_out, nullptr, (float*)d_out, nullptr,
        M, 1024, 2048, 2048, 2048, 1024, 0);
}
